// TransMIL_35021163332134
// MI455X (gfx1250) — compile-verified, matrix operand order checked by reading, not run on hardware
//
#include <hip/hip_runtime.h>
#include <hip/hip_bf16.h>

typedef __bf16 bf16_t;
typedef __attribute__((ext_vector_type(16))) __bf16 v16bf;
typedef __attribute__((ext_vector_type(8)))  float  v8f;

#define N0    20000
#define K0    1024
#define DIM   512
#define HS    142
#define NSQ   20164   /* HS*HS */
#define ADD   164
#define NT    20165   /* NSQ + 1 (cls) */
#define PN    20224   /* padded seq len, 79*256 */
#define PADR  59
#define HEADS 8
#define DH    64
#define LM    256     /* landmarks */
#define LCH   79      /* rows averaged per landmark */
#define EPSLN 1e-5f

// ---------------------------------------------------------------------------
// f32 -> bf16 cast
// ---------------------------------------------------------------------------
__global__ void cast_bf16_k(const float* __restrict__ s, bf16_t* __restrict__ d, long n) {
    long i  = (long)blockIdx.x * blockDim.x + threadIdx.x;
    long st = (long)gridDim.x * blockDim.x;
    for (; i < n; i += st) d[i] = (bf16_t)s[i];
}

// ---------------------------------------------------------------------------
// Generic WMMA GEMM:  C[b] = act( scale * A[b] @ B[b]^T + bias + diagc*I )
// A: (Mr x K) bf16 row-major (lda), B: (N x K) bf16 row-major (ldb).
// Wave tile: 32x64 (2 A-frags x 4 B-frags -> 8 WMMAs per k-step).
// Block: 8 waves as 2(M) x 4(N)  ->  64 x 256 block tile.
// Mr multiple of 32, N multiple of 256, K multiple of 32 (all uses satisfy).
// ---------------------------------------------------------------------------
__global__ void gemm_wmma_k(const bf16_t* __restrict__ A, const bf16_t* __restrict__ B,
                            float* __restrict__ Cf, bf16_t* __restrict__ Cb,
                            const float* __restrict__ bias,
                            int Mr, int N, int K, int lda, int ldb, int ldc,
                            long sA, long sB, long sC,
                            float scale, float diagc, int relu) {
    int wave = threadIdx.x >> 5;    // 0..7
    int lane = threadIdx.x & 31;
    int wm = wave >> 2;             // 0..1
    int wn = wave & 3;              // 0..3
    int row0 = blockIdx.y * 64 + wm * 32;
    int col0 = blockIdx.x * 256 + wn * 64;
    if (row0 >= Mr || col0 >= N) return;   // wave-uniform: EXEC stays all-ones
    int bz = blockIdx.z;
    int lr = lane & 15, lh = lane >> 4;
    const bf16_t* ap0 = A + (long)bz * sA + (long)(row0 + lr) * lda + lh * 16;
    const bf16_t* ap1 = ap0 + (long)16 * lda;
    const bf16_t* bp  = B + (long)bz * sB + (long)(col0 + lr) * ldb + lh * 16;
    long bstep = (long)16 * ldb;
    v8f acc[8] = {};                 // acc[mi*4 + ni]
    for (int k = 0; k < K; k += 32) {
        // prefetch one k-block (64B) ahead on each stream
        __builtin_prefetch(ap0 + k + 32, 0, 1);
        __builtin_prefetch(ap1 + k + 32, 0, 1);
        __builtin_prefetch(bp + k + 32, 0, 1);
        v16bf a0 = *(const v16bf*)(ap0 + k);
        v16bf a1 = *(const v16bf*)(ap1 + k);
#pragma unroll
        for (int ni = 0; ni < 4; ++ni) {
            v16bf b = *(const v16bf*)(bp + (long)ni * bstep + k);
            acc[ni]     = __builtin_amdgcn_wmma_f32_16x16x32_bf16(false, a0, false, b,
                                                                  (short)0, acc[ni], false, false);
            acc[4 + ni] = __builtin_amdgcn_wmma_f32_16x16x32_bf16(false, a1, false, b,
                                                                  (short)0, acc[4 + ni], false, false);
        }
    }
    // C layout per tile: lane<16 -> N=lane, M=vg ; lane>=16 -> N=lane-16, M=vg+8
#pragma unroll
    for (int mi = 0; mi < 2; ++mi) {
        int rb = row0 + mi * 16 + lh * 8;
#pragma unroll
        for (int ni = 0; ni < 4; ++ni) {
            int   cc = col0 + ni * 16 + lr;
            float bb = bias ? bias[cc] : 0.f;
            v8f   a  = acc[mi * 4 + ni];
#pragma unroll
            for (int g = 0; g < 8; ++g) {
                int   rr = rb + g;
                float v  = scale * a[g] + bb + ((rr == cc) ? diagc : 0.f);
                if (relu) v = fmaxf(v, 0.f);
                long off = (long)bz * sC + (long)rr * ldc + cc;
                if (Cf) Cf[off] = v;
                if (Cb) Cb[off] = (bf16_t)v;
            }
        }
    }
}

// ---------------------------------------------------------------------------
// h row 0 = cls token; duplicate first ADD feature rows to the tail
// ---------------------------------------------------------------------------
__global__ void build_h_k(float* __restrict__ h, const float* __restrict__ cls) {
    int tot = (1 + ADD) * DIM;
    for (int idx = blockIdx.x * blockDim.x + threadIdx.x; idx < tot;
         idx += gridDim.x * blockDim.x) {
        if (idx < DIM) {
            h[idx] = cls[idx];
        } else {
            int k = idx - DIM;
            int j = k >> 9, c = k & 511;
            h[(long)(1 + N0 + j) * DIM + c] = h[(long)(1 + j) * DIM + c];
        }
    }
}

// ---------------------------------------------------------------------------
// LayerNorm of h rows -> bf16, written at row+PADR (rows 0..PADR-1 zeroed)
// ---------------------------------------------------------------------------
__global__ void ln_pad_k(const float* __restrict__ h, const float* __restrict__ g,
                         const float* __restrict__ b, bf16_t* __restrict__ out) {
    int r = blockIdx.x, t = threadIdx.x;       // block 256
    bf16_t* orow = out + (long)r * DIM;
    if (r < PADR) { orow[t] = (bf16_t)0.f; orow[t + 256] = (bf16_t)0.f; return; }
    const float* x = h + (long)(r - PADR) * DIM;
    __shared__ float red[256];
    float x0 = x[t], x1 = x[t + 256];
    red[t] = x0 + x1; __syncthreads();
    for (int s = 128; s > 0; s >>= 1) { if (t < s) red[t] += red[t + s]; __syncthreads(); }
    float mu = red[0] * (1.f / DIM); __syncthreads();
    float d0 = x0 - mu, d1 = x1 - mu;
    red[t] = d0 * d0 + d1 * d1; __syncthreads();
    for (int s = 128; s > 0; s >>= 1) { if (t < s) red[t] += red[t + s]; __syncthreads(); }
    float w = rsqrtf(red[0] * (1.f / DIM) + EPSLN);
    orow[t]       = (bf16_t)(d0 * w * g[t]       + b[t]);
    orow[t + 256] = (bf16_t)(d1 * w * g[t + 256] + b[t + 256]);
}

// ---------------------------------------------------------------------------
// Landmarks: ql/kl[h][m][d] = mean_j q/k[m*79+j][h][d]   (unscaled q)
// ---------------------------------------------------------------------------
__global__ void landmark_k(const bf16_t* __restrict__ qkv,
                           float* __restrict__ ql, float* __restrict__ kl) {
    int m = blockIdx.x, h = blockIdx.y, d = threadIdx.x;  // block 64
    const bf16_t* base = qkv + (long)(m * LCH) * 1536 + h * DH + d;
    float sq = 0.f, sk = 0.f;
    for (int j = 0; j < LCH; ++j) {
        sq += (float)base[(long)j * 1536];
        sk += (float)base[(long)j * 1536 + DIM];
    }
    int o = (h * LM + m) * DH + d;
    ql[o] = sq * (1.f / LCH);
    kl[o] = sk * (1.f / LCH);
}

// ---------------------------------------------------------------------------
// a2 = softmax(0.125 * ql @ kl^T) per head, f32 + bf16 outputs
// ---------------------------------------------------------------------------
__global__ void a2_softmax_k(const float* __restrict__ ql, const float* __restrict__ kl,
                             float* __restrict__ a2f, bf16_t* __restrict__ a2b) {
    int i = blockIdx.x, h = blockIdx.y, t = threadIdx.x;  // block 256
    __shared__ float q[DH];
    __shared__ float red[256];
    if (t < DH) q[t] = ql[(h * LM + i) * DH + t];
    __syncthreads();
    const float* kr = kl + (h * LM + t) * DH;
    float s = 0.f;
#pragma unroll 8
    for (int d = 0; d < DH; ++d) s += q[d] * kr[d];
    s *= 0.125f;
    red[t] = s; __syncthreads();
    for (int w = 128; w > 0; w >>= 1) { if (t < w) red[t] = fmaxf(red[t], red[t + w]); __syncthreads(); }
    float gm = red[0]; __syncthreads();
    float e = __expf(s - gm);
    red[t] = e; __syncthreads();
    for (int w = 128; w > 0; w >>= 1) { if (t < w) red[t] += red[t + w]; __syncthreads(); }
    float p = e / red[0];
    long o = (long)h * LM * LM + (long)i * LM + t;
    a2f[o] = p;
    a2b[o] = (bf16_t)p;
}

// ---------------------------------------------------------------------------
// scal[0] = 1 / ( max(rowsum |a2|) * max(colsum |a2|) )   (a2 >= 0)
// ---------------------------------------------------------------------------
__global__ void a2_norm_k(const float* __restrict__ a2f, float* __restrict__ scal) {
    int t = threadIdx.x;  // single block of 256
    __shared__ float red[256];
    float rmax = 0.f;
    for (int r = t; r < HEADS * LM; r += 256) {
        const float* p = a2f + (long)r * LM;
        float s = 0.f;
        for (int c = 0; c < LM; ++c) s += p[c];
        rmax = fmaxf(rmax, s);
    }
    float cmax = 0.f;
    for (int id = t; id < HEADS * LM; id += 256) {
        int hh = id >> 8, c = id & 255;
        const float* p = a2f + (long)hh * LM * LM + c;
        float s = 0.f;
        for (int r = 0; r < LM; ++r) s += p[(long)r * LM];
        cmax = fmaxf(cmax, s);
    }
    red[t] = rmax; __syncthreads();
    for (int w = 128; w > 0; w >>= 1) { if (t < w) red[t] = fmaxf(red[t], red[t + w]); __syncthreads(); }
    rmax = red[0]; __syncthreads();
    red[t] = cmax; __syncthreads();
    for (int w = 128; w > 0; w >>= 1) { if (t < w) red[t] = fmaxf(red[t], red[t + w]); __syncthreads(); }
    if (t == 0) scal[0] = 1.f / (rmax * red[0]);
}

// z0 = a2^T * scal
__global__ void z0_k(const float* __restrict__ a2f, const float* __restrict__ scal,
                     bf16_t* __restrict__ z) {
    long n = (long)HEADS * LM * LM;
    float sc = scal[0];
    for (long idx = (long)blockIdx.x * blockDim.x + threadIdx.x; idx < n;
         idx += (long)gridDim.x * blockDim.x) {
        long h = idx >> 16;
        int  r = (int)((idx >> 8) & 255), c = (int)(idx & 255);
        z[idx] = (bf16_t)(a2f[(h << 16) + ((long)c << 8) + r] * sc);
    }
}

// batched 256x256 bf16 transpose (so GEMM B-fragments stay contiguous)
__global__ void tr256_k(const bf16_t* __restrict__ in, bf16_t* __restrict__ out) {
    int h = blockIdx.z;
    int r = blockIdx.x * 16 + threadIdx.y;
    int c = blockIdx.y * 16 + threadIdx.x;
    out[(long)h * 65536 + (long)c * 256 + r] = in[(long)h * 65536 + (long)r * 256 + c];
}

// out = s*in + dc*I   (bf16)
__global__ void diagscale_k(const bf16_t* __restrict__ in, bf16_t* __restrict__ out,
                            float s, float dc) {
    long n = (long)HEADS * 65536;
    for (long idx = (long)blockIdx.x * blockDim.x + threadIdx.x; idx < n;
         idx += (long)gridDim.x * blockDim.x) {
        int r = (int)((idx >> 8) & 255), c = (int)(idx & 255);
        out[idx] = (bf16_t)(s * (float)in[idx] + ((r == c) ? dc : 0.f));
    }
}

// ---------------------------------------------------------------------------
// av = softmax(0.125 * ql @ k^T) @ v  -- flash-style streaming over PN keys
// grid (LM, HEADS), block 256
// ---------------------------------------------------------------------------
__global__ void a3v_k(const bf16_t* __restrict__ qkv, const float* __restrict__ ql,
                      float* __restrict__ av) {
    int j = blockIdx.x, h = blockIdx.y, t = threadIdx.x;
    __shared__ float qlr[DH];
    __shared__ float red[256];
    __shared__ float sv[256];
    if (t < DH) qlr[t] = ql[(h * LM + j) * DH + t];
    __syncthreads();
    // phase 1: per-thread online max/sum over strided keys
    float m = -1e30f, ssum = 0.f;
    for (int i0 = 0; i0 < PN; i0 += 256) {
        const bf16_t* kr = qkv + (long)(i0 + t) * 1536 + DIM + h * DH;
        float d = 0.f;
#pragma unroll 8
        for (int c = 0; c < DH; ++c) d += qlr[c] * (float)kr[c];
        d *= 0.125f;
        float nm = fmaxf(m, d);
        ssum = ssum * __expf(m - nm) + __expf(d - nm);
        m = nm;
    }
    red[t] = m; __syncthreads();
    for (int w = 128; w > 0; w >>= 1) { if (t < w) red[t] = fmaxf(red[t], red[t + w]); __syncthreads(); }
    float gm = red[0]; __syncthreads();
    red[t] = ssum * __expf(m - gm); __syncthreads();
    for (int w = 128; w > 0; w >>= 1) { if (t < w) red[t] += red[t + w]; __syncthreads(); }
    float gs = red[0]; __syncthreads();
    // phase 2: weighted sum of v
    int dd = t & 63, qr = t >> 6;
    float acc = 0.f;
    for (int i0 = 0; i0 < PN; i0 += 256) {
        const bf16_t* kr = qkv + (long)(i0 + t) * 1536 + DIM + h * DH;
        float d = 0.f;
#pragma unroll 8
        for (int c = 0; c < DH; ++c) d += qlr[c] * (float)kr[c];
        sv[t] = __expf(d * 0.125f - gm);
        __syncthreads();
        const bf16_t* vb = qkv + (long)(i0 + qr * 64) * 1536 + 1024 + h * DH + dd;
#pragma unroll 8
        for (int c = 0; c < 64; ++c) acc += sv[qr * 64 + c] * (float)vb[(long)c * 1536];
        __syncthreads();
    }
    red[t] = acc; __syncthreads();
    if (t < 64)
        av[(h * LM + j) * DH + t] =
            (red[t] + red[t + 64] + red[t + 128] + red[t + 192]) / gs;
}

// z2 = pinv(a2) @ av  :  grid (LM, HEADS), block 64
__global__ void z2_k(const bf16_t* __restrict__ z, const float* __restrict__ av,
                     float* __restrict__ z2) {
    int j = blockIdx.x, h = blockIdx.y, t = threadIdx.x;
    __shared__ float zr[LM];
    for (int m = t; m < LM; m += 64) zr[m] = (float)z[(long)h * 65536 + (long)j * 256 + m];
    __syncthreads();
    float a = 0.f;
#pragma unroll 8
    for (int m = 0; m < LM; ++m) a += zr[m] * av[(h * LM + m) * DH + t];
    z2[(h * LM + j) * DH + t] = a;
}

// ---------------------------------------------------------------------------
// out_row = softmax(0.125 * q_i @ kl^T) @ z2 ; grid (PN, HEADS), block 256
// ---------------------------------------------------------------------------
__global__ void a1_out_k(const bf16_t* __restrict__ qkv, const float* __restrict__ kl,
                         const float* __restrict__ z2, float* __restrict__ attn) {
    int i = blockIdx.x, h = blockIdx.y, t = threadIdx.x;
    __shared__ float q[DH];
    __shared__ float sv[256];
    __shared__ float red[256];
    if (t < DH) q[t] = 0.125f * (float)qkv[(long)i * 1536 + h * DH + t];
    __syncthreads();
    const float* kr = kl + (h * LM + t) * DH;
    float s = 0.f;
#pragma unroll 8
    for (int d = 0; d < DH; ++d) s += q[d] * kr[d];
    red[t] = s; __syncthreads();
    for (int w = 128; w > 0; w >>= 1) { if (t < w) red[t] = fmaxf(red[t], red[t + w]); __syncthreads(); }
    float gm = red[0]; __syncthreads();
    float e = __expf(s - gm);
    sv[t] = e;
    red[t] = e; __syncthreads();
    for (int w = 128; w > 0; w >>= 1) { if (t < w) red[t] += red[t + w]; __syncthreads(); }
    float inv = 1.f / red[0];
    __syncthreads();
    int dd = t & 63, qq = t >> 6;
    const float* zp = z2 + ((long)h * LM + qq * 64) * DH + dd;
    float a = 0.f;
#pragma unroll 8
    for (int c = 0; c < 64; ++c) a += sv[qq * 64 + c] * zp[(long)c * DH];
    red[t] = a; __syncthreads();
    if (t < 64)
        attn[(long)i * DIM + h * DH + t] =
            (red[t] + red[t + 64] + red[t + 128] + red[t + 192]) * inv;
}

// depthwise 33-tap residual conv on v along the sequence, add + cast to bf16
__global__ void resconv_k(const bf16_t* __restrict__ qkv, const float* __restrict__ rw,
                          float* __restrict__ attn, bf16_t* __restrict__ attnb) {
    long n = (long)PN * DIM;
    for (long idx = (long)blockIdx.x * blockDim.x + threadIdx.x; idx < n;
         idx += (long)gridDim.x * blockDim.x) {
        int  hd = (int)(idx & 511);
        long i  = idx >> 9;
        int  hh = hd >> 6;
        float a = attn[idx];
        const float* w = rw + hh * 33;
        for (int t2 = 0; t2 < 33; ++t2) {
            long ii = i + t2 - 16;
            if (ii >= 0 && ii < PN) a += w[t2] * (float)qkv[ii * 1536 + 1024 + hd];
        }
        attn[idx]  = a;
        attnb[idx] = (bf16_t)a;
    }
}

// residual add (drop the PADR leading rows of proj)
__global__ void resid_k(float* __restrict__ h, const float* __restrict__ proj) {
    long n = (long)NT * DIM;
    for (long idx = (long)blockIdx.x * blockDim.x + threadIdx.x; idx < n;
         idx += (long)gridDim.x * blockDim.x)
        h[idx] += proj[idx + (long)PADR * DIM];
}

__global__ void copy_row_k(float* __restrict__ dst, const float* __restrict__ src) {
    int t = threadIdx.x;
    dst[t] = src[t];
    dst[t + 256] = src[t + 256];
}

// PPEG: y = f + dw7(f)+b7 + dw5(f)+b5 + dw3(f)+b3 over the 142x142 grid
__global__ void ppeg_k(const float* __restrict__ h,
                       const float* __restrict__ w7, const float* __restrict__ b7,
                       const float* __restrict__ w5, const float* __restrict__ b5,
                       const float* __restrict__ w3, const float* __restrict__ b3,
                       float* __restrict__ h2) {
    long n = (long)NSQ * DIM;
    for (long idx = (long)blockIdx.x * blockDim.x + threadIdx.x; idx < n;
         idx += (long)gridDim.x * blockDim.x) {
        int  c   = (int)(idx & 511);
        long pnl = idx >> 9;
        int  hh = (int)(pnl / HS), ww = (int)(pnl % HS);
        const float* base = h + DIM + c;           // feat[n][c] = h[(1+n)*512+c]
        float acc = base[pnl * DIM] + b7[c] + b5[c] + b3[c];
        const float* W7 = w7 + c * 49;
        for (int kh = 0; kh < 7; ++kh) {
            int y = hh + kh - 3; if ((unsigned)y >= HS) continue;
            for (int kw = 0; kw < 7; ++kw) {
                int x = ww + kw - 3; if ((unsigned)x >= HS) continue;
                acc += W7[kh * 7 + kw] * base[((long)y * HS + x) * DIM];
            }
        }
        const float* W5 = w5 + c * 25;
        for (int kh = 0; kh < 5; ++kh) {
            int y = hh + kh - 2; if ((unsigned)y >= HS) continue;
            for (int kw = 0; kw < 5; ++kw) {
                int x = ww + kw - 2; if ((unsigned)x >= HS) continue;
                acc += W5[kh * 5 + kw] * base[((long)y * HS + x) * DIM];
            }
        }
        const float* W3 = w3 + c * 9;
        for (int kh = 0; kh < 3; ++kh) {
            int y = hh + kh - 1; if ((unsigned)y >= HS) continue;
            for (int kw = 0; kw < 3; ++kw) {
                int x = ww + kw - 1; if ((unsigned)x >= HS) continue;
                acc += W3[kh * 3 + kw] * base[((long)y * HS + x) * DIM];
            }
        }
        h2[DIM + idx] = acc;
    }
}

// final: out = LN(h2 row0) @ fc2_w^T + fc2_b
__global__ void final_k(const float* __restrict__ h, const float* __restrict__ g,
                        const float* __restrict__ b, const float* __restrict__ w,
                        const float* __restrict__ fb, float* __restrict__ out) {
    int t = threadIdx.x;  // 256
    __shared__ float red[256];
    float x0 = h[t], x1 = h[t + 256];
    red[t] = x0 + x1; __syncthreads();
    for (int s = 128; s > 0; s >>= 1) { if (t < s) red[t] += red[t + s]; __syncthreads(); }
    float mu = red[0] * (1.f / DIM); __syncthreads();
    float d0 = x0 - mu, d1 = x1 - mu;
    red[t] = d0 * d0 + d1 * d1; __syncthreads();
    for (int s = 128; s > 0; s >>= 1) { if (t < s) red[t] += red[t + s]; __syncthreads(); }
    float iw = rsqrtf(red[0] * (1.f / DIM) + EPSLN); __syncthreads();
    float c0 = d0 * iw * g[t] + b[t], c1 = d1 * iw * g[t + 256] + b[t + 256];
    for (int o = 0; o < 2; ++o) {
        red[t] = c0 * w[o * DIM + t] + c1 * w[o * DIM + t + 256]; __syncthreads();
        for (int s = 128; s > 0; s >>= 1) { if (t < s) red[t] += red[t + s]; __syncthreads(); }
        if (t == 0) out[o] = red[0] + fb[o];
        __syncthreads();
    }
}

// ---------------------------------------------------------------------------
// Host orchestration
// ---------------------------------------------------------------------------
static inline size_t alignup256(size_t x) { return (x + 255) & ~(size_t)255; }

extern "C" void kernel_launch(void* const* d_in, const int* in_sizes, int n_in,
                              void* d_out, int out_size, void* d_ws, size_t ws_size,
                              hipStream_t stream) {
    (void)in_sizes; (void)n_in; (void)out_size; (void)ws_size;
    const float* x      = (const float*)d_in[0];
    const float* fc1w   = (const float*)d_in[1];
    const float* fc1b   = (const float*)d_in[2];
    const float* clstok = (const float*)d_in[3];
    const float* l1ng   = (const float*)d_in[4];
    const float* l1nb   = (const float*)d_in[5];
    const float* l1qkvw = (const float*)d_in[6];
    const float* l1outw = (const float*)d_in[7];
    const float* l1outb = (const float*)d_in[8];
    const float* l1resw = (const float*)d_in[9];
    const float* w7 = (const float*)d_in[10]; const float* b7 = (const float*)d_in[11];
    const float* w5 = (const float*)d_in[12]; const float* b5 = (const float*)d_in[13];
    const float* w3 = (const float*)d_in[14]; const float* b3 = (const float*)d_in[15];
    const float* l2ng = (const float*)d_in[16]; const float* l2nb = (const float*)d_in[17];
    const float* l2qkvw = (const float*)d_in[18];
    const float* l2outw = (const float*)d_in[19];
    const float* l2outb = (const float*)d_in[20];
    const float* l2resw = (const float*)d_in[21];
    const float* ngf = (const float*)d_in[22]; const float* nbf = (const float*)d_in[23];
    const float* fc2w = (const float*)d_in[24]; const float* fc2b = (const float*)d_in[25];
    float* outp = (float*)d_out;

    char* p = (char*)d_ws;
    auto take = [&](size_t bytes) { char* r = p; p += alignup256(bytes); return r; };
    bf16_t* xbf   = (bf16_t*)take((size_t)N0 * K0 * 2);
    bf16_t* wfc1  = (bf16_t*)take((size_t)DIM * K0 * 2);
    bf16_t* wq1   = (bf16_t*)take((size_t)1536 * DIM * 2);
    bf16_t* wq2   = (bf16_t*)take((size_t)1536 * DIM * 2);
    bf16_t* wo1   = (bf16_t*)take((size_t)DIM * DIM * 2);
    bf16_t* wo2   = (bf16_t*)take((size_t)DIM * DIM * 2);
    float*  hbuf  = (float*)take((size_t)NT * DIM * 4);
    float*  h2buf = (float*)take((size_t)NT * DIM * 4);
    bf16_t* hnbf  = (bf16_t*)take((size_t)PN * DIM * 2);
    bf16_t* qkvbf = (bf16_t*)take((size_t)PN * 1536 * 2);
    float*  ql    = (float*)take((size_t)HEADS * LM * DH * 4);
    float*  kl    = (float*)take((size_t)HEADS * LM * DH * 4);
    float*  a2f   = (float*)take((size_t)HEADS * LM * LM * 4);
    bf16_t* a2b   = (bf16_t*)take((size_t)HEADS * LM * LM * 2);
    bf16_t* zA    = (bf16_t*)take((size_t)HEADS * LM * LM * 2);
    bf16_t* zB    = (bf16_t*)take((size_t)HEADS * LM * LM * 2);
    bf16_t* xz    = (bf16_t*)take((size_t)HEADS * LM * LM * 2);
    bf16_t* w1b   = (bf16_t*)take((size_t)HEADS * LM * LM * 2);
    bf16_t* w2b   = (bf16_t*)take((size_t)HEADS * LM * LM * 2);
    bf16_t* w3b   = (bf16_t*)take((size_t)HEADS * LM * LM * 2);
    bf16_t* tb    = (bf16_t*)take((size_t)HEADS * LM * LM * 2);
    float*  scal  = (float*)take(256);
    float*  av    = (float*)take((size_t)HEADS * LM * DH * 4);
    float*  z2b   = (float*)take((size_t)HEADS * LM * DH * 4);
    float*  attnf = (float*)take((size_t)PN * DIM * 4);
    bf16_t* attnb = (bf16_t*)take((size_t)PN * DIM * 2);

    auto cast = [&](const float* s, bf16_t* d, long n) {
        int blocks = (int)((n + 255) / 256);
        if (blocks > 4096) blocks = 4096;
        cast_bf16_k<<<dim3(blocks), dim3(256), 0, stream>>>(s, d, n);
    };
    auto gemm = [&](const bf16_t* A, const bf16_t* B, float* Cf, bf16_t* Cb,
                    const float* bias, int Mr, int N, int K, int lda, int ldb, int ldc,
                    int batch, long sA, long sB, long sC, float sc, float dc, int relu) {
        dim3 grid((N + 255) / 256, (Mr + 63) / 64, batch);
        gemm_wmma_k<<<grid, dim3(256), 0, stream>>>(A, B, Cf, Cb, bias, Mr, N, K,
                                                    lda, ldb, ldc, sA, sB, sC, sc, dc, relu);
    };

    auto nystrom = [&](float* hres, const float* lng, const float* lnb,
                       const bf16_t* wqkv, const bf16_t* wout, const float* outb,
                       const float* resw) {
        ln_pad_k<<<dim3(PN), dim3(256), 0, stream>>>(hres, lng, lnb, hnbf);
        gemm(hnbf, wqkv, nullptr, qkvbf, nullptr, PN, 1536, DIM, DIM, DIM, 1536,
             1, 0, 0, 0, 1.f, 0.f, 0);
        landmark_k<<<dim3(LM, HEADS), dim3(64), 0, stream>>>(qkvbf, ql, kl);
        a2_softmax_k<<<dim3(LM, HEADS), dim3(256), 0, stream>>>(ql, kl, a2f, a2b);
        a2_norm_k<<<dim3(1), dim3(256), 0, stream>>>(a2f, scal);
        z0_k<<<dim3(512), dim3(256), 0, stream>>>(a2f, scal, zA);
        bf16_t* zc = zA; bf16_t* zn = zB;
        const long S = (long)LM * LM;
        for (int it = 0; it < 6; ++it) {
            tr256_k<<<dim3(16, 16, HEADS), dim3(16, 16), 0, stream>>>(zc, tb);
            gemm(a2b, tb, nullptr, xz, nullptr, LM, LM, LM, LM, LM, LM, HEADS,
                 S, S, S, 1.f, 0.f, 0);
            diagscale_k<<<dim3(512), dim3(256), 0, stream>>>(xz, w1b, -1.f, 7.f);
            tr256_k<<<dim3(16, 16, HEADS), dim3(16, 16), 0, stream>>>(w1b, tb);
            gemm(xz, tb, nullptr, w2b, nullptr, LM, LM, LM, LM, LM, LM, HEADS,
                 S, S, S, -1.f, 15.f, 0);
            tr256_k<<<dim3(16, 16, HEADS), dim3(16, 16), 0, stream>>>(w2b, tb);
            gemm(xz, tb, nullptr, w3b, nullptr, LM, LM, LM, LM, LM, LM, HEADS,
                 S, S, S, -1.f, 13.f, 0);
            tr256_k<<<dim3(16, 16, HEADS), dim3(16, 16), 0, stream>>>(w3b, tb);
            gemm(zc, tb, nullptr, zn, nullptr, LM, LM, LM, LM, LM, LM, HEADS,
                 S, S, S, 0.25f, 0.f, 0);
            bf16_t* tmp = zc; zc = zn; zn = tmp;
        }
        a3v_k<<<dim3(LM, HEADS), dim3(256), 0, stream>>>(qkvbf, ql, av);
        z2_k<<<dim3(LM, HEADS), dim3(64), 0, stream>>>(zc, av, z2b);
        a1_out_k<<<dim3(PN, HEADS), dim3(256), 0, stream>>>(qkvbf, kl, z2b, attnf);
        resconv_k<<<dim3(4096), dim3(256), 0, stream>>>(qkvbf, resw, attnf, attnb);
        // out projection writes f32 back into attnf (attnb holds the A operand)
        gemm(attnb, wout, attnf, nullptr, outb, PN, DIM, DIM, DIM, DIM, DIM,
             1, 0, 0, 0, 1.f, 0.f, 0);
        resid_k<<<dim3(4096), dim3(256), 0, stream>>>(hres, attnf);
    };

    // ---- pipeline ----
    cast(x,      xbf,  (long)N0 * K0);
    cast(fc1w,   wfc1, (long)DIM * K0);
    cast(l1qkvw, wq1,  (long)1536 * DIM);
    cast(l2qkvw, wq2,  (long)1536 * DIM);
    cast(l1outw, wo1,  (long)DIM * DIM);
    cast(l2outw, wo2,  (long)DIM * DIM);

    // fc1 + relu, written at h rows 1..20000
    gemm(xbf, wfc1, hbuf + DIM, nullptr, fc1b, N0, DIM, K0, K0, K0, DIM,
         1, 0, 0, 0, 1.f, 0.f, 1);
    build_h_k<<<dim3(330), dim3(256), 0, stream>>>(hbuf, clstok);

    nystrom(hbuf, l1ng, l1nb, wq1, wo1, l1outb, l1resw);

    copy_row_k<<<dim3(1), dim3(256), 0, stream>>>(h2buf, hbuf);
    ppeg_k<<<dim3(8192), dim3(256), 0, stream>>>(hbuf, w7, b7, w5, b5, w3, b3, h2buf);

    nystrom(h2buf, l2ng, l2nb, wq2, wo2, l2outb, l2resw);

    final_k<<<dim3(1), dim3(256), 0, stream>>>(h2buf, ngf, nbf, fc2w, fc2b, outp);
}
